// CascadeStage_62010737819705
// MI455X (gfx1250) — compile-verified, matrix operand order checked by reading, not run on hardware
//
#include <hip/hip_runtime.h>
#include <math.h>

// ---------------- problem constants ----------------
#define BB   4
#define NN_  1024
#define DD_  512
#define NH_  8
#define DK_  64
#define DFF_ 2048
#define DRAW_ 32
#define TMAX_ 2
#define BNROWS (BB*NN_)            // 4096

typedef __bf16 bf16x16 __attribute__((ext_vector_type(16)));
typedef float  f32x8   __attribute__((ext_vector_type(8)));
typedef unsigned int u32x4 __attribute__((ext_vector_type(4)));
typedef int i32x4 __attribute__((ext_vector_type(4)));
typedef int i32x8 __attribute__((ext_vector_type(8)));

// ---------------- helpers ----------------
__device__ __forceinline__ unsigned short f2bf(float f) {
    unsigned u = __builtin_bit_cast(unsigned, f);
    unsigned r = u + 0x7FFFu + ((u >> 16) & 1u);
    return (unsigned short)(r >> 16);
}
__device__ __forceinline__ float softplusf(float x) {
    return (x > 20.f) ? x : log1pf(expf(x));
}

union FragU { u32x4 q[2]; bf16x16 v; };

// Fragment per documented 16-bit 16x32 A layout (B symmetric):
// lane L: row r = L&15, chunk c = L>>4; elems 0..7 = K[c*8 .. c*8+7],
// elems 8..15 = K[16+c*8 .. 16+c*8+7]  (two contiguous 16-byte loads).
__device__ __forceinline__ bf16x16 load_frag(const unsigned short* base, int ld, int lane) {
    int r = lane & 15, c = lane >> 4;
    const unsigned short* p = base + (size_t)r * ld + c * 8;
    FragU f;
    f.q[0] = *(const u32x4*)(p);
    f.q[1] = *(const u32x4*)(p + 16);
    return f.v;
}

__device__ __forceinline__ unsigned ldsOff(const void* p) {
    return (unsigned)(unsigned long long)p;   // flat LDS address: offset in low 32 bits
}

// Build D# (groups 0/1) for a 2D bf16 tile [rows x cols] with row stride ld (elements)
// and issue TENSOR_LOAD_TO_LDS (clang-23 / therock 6-arg builtin).  Tracked by TENSORcnt.
__device__ __forceinline__ void tdm_load_tile(const unsigned short* gptr, unsigned lds_off,
                                              int rows, int cols, int ld)
{
    unsigned long long ga = (unsigned long long)(const void*)gptr;
    u32x4 g0;
    g0[0] = 1u;                                    // count=1, user descriptor, no gather
    g0[1] = lds_off;                               // lds_addr (bytes)
    g0[2] = (unsigned)(ga & 0xFFFFFFFFull);        // global_addr[31:0]
    g0[3] = (unsigned)(ga >> 32) | 0x80000000u;    // global_addr[56:32] | type=2 ("image")
    i32x8 g1;
    unsigned ucols = (unsigned)cols, urows = (unsigned)rows, uld = (unsigned)ld;
    g1[0] = 0x00010000;                            // workgroup_mask=0, data_size=1 (2 bytes)
    g1[1] = (int)(ucols << 16);                    // tensor_dim0[15:0]
    g1[2] = (int)((ucols >> 16) | (urows << 16));  // tensor_dim0[31:16] | tensor_dim1[15:0]
    g1[3] = (int)((urows >> 16) | (ucols << 16));  // tensor_dim1[31:16] | tile_dim0
    g1[4] = (int)(urows & 0xFFFFu);                // tile_dim1 | tile_dim2=0
    g1[5] = (int)uld;                              // tensor_dim0_stride[31:0]
    g1[6] = 0;                                     // stride[47:32] | dim1_stride lo
    g1[7] = 0;
    i32x4 z4 = {};
    i32x8 z8 = {};
    __builtin_amdgcn_tensor_load_to_lds(g0, g1, z4, z4, z8, 0);
}

// ---------------- direct-load WMMA GEMM (used for small-K score GEMMs) ----------------
// C[M,N] = alpha * A[M,K] * B[N,K]^T + bias[N];  A,B bf16 K-contiguous.
__global__ __launch_bounds__(128) void gemm_bt_kernel(
    const unsigned short* __restrict__ A, const unsigned short* __restrict__ B,
    float* __restrict__ C, const float* __restrict__ bias,
    int M, int N, int K, int lda, int ldb, int ldc, float alpha)
{
    const int lane = threadIdx.x & 31;
    const int wave = threadIdx.x >> 5;           // 0..3
    const int row0 = blockIdx.y * 64 + (wave >> 1) * 32;
    const int col0 = blockIdx.x * 64 + (wave & 1) * 32;
    (void)M; (void)N;

    f32x8 acc00 = {}; f32x8 acc01 = {}; f32x8 acc10 = {}; f32x8 acc11 = {};

    for (int k0 = 0; k0 < K; k0 += 32) {
        if (k0 + 32 < K) {
            __builtin_prefetch(A + (size_t)row0 * lda + k0 + 32, 0, 1);
            __builtin_prefetch(B + (size_t)col0 * ldb + k0 + 32, 0, 1);
        }
        bf16x16 a0 = load_frag(A + (size_t)(row0     ) * lda + k0, lda, lane);
        bf16x16 a1 = load_frag(A + (size_t)(row0 + 16) * lda + k0, lda, lane);
        bf16x16 b0 = load_frag(B + (size_t)(col0     ) * ldb + k0, ldb, lane);
        bf16x16 b1 = load_frag(B + (size_t)(col0 + 16) * ldb + k0, ldb, lane);
        acc00 = __builtin_amdgcn_wmma_f32_16x16x32_bf16(false, a0, false, b0, (short)0, acc00, false, false);
        acc01 = __builtin_amdgcn_wmma_f32_16x16x32_bf16(false, a0, false, b1, (short)0, acc01, false, false);
        acc10 = __builtin_amdgcn_wmma_f32_16x16x32_bf16(false, a1, false, b0, (short)0, acc10, false, false);
        acc11 = __builtin_amdgcn_wmma_f32_16x16x32_bf16(false, a1, false, b1, (short)0, acc11, false, false);
    }

    const int crow = (lane >> 4) * 8;
    const int ccol = lane & 15;
    const float bc0 = bias ? bias[col0 + ccol]      : 0.f;
    const float bc1 = bias ? bias[col0 + 16 + ccol] : 0.f;
    #pragma unroll
    for (int j = 0; j < 8; ++j) {
        int r0 = row0 + crow + j;
        int r1 = r0 + 16;
        C[(size_t)r0 * ldc + col0 + ccol]      = alpha * acc00[j] + bc0;
        C[(size_t)r0 * ldc + col0 + 16 + ccol] = alpha * acc01[j] + bc1;
        C[(size_t)r1 * ldc + col0 + ccol]      = alpha * acc10[j] + bc0;
        C[(size_t)r1 * ldc + col0 + 16 + ccol] = alpha * acc11[j] + bc1;
    }
}

// ---------------- TDM + LDS double-buffered WMMA GEMM (large-K GEMMs) ----------------
// One TENSOR_LOAD_TO_LDS per 64x32 tile per block; compute overlaps next-tile DMA.
__global__ __launch_bounds__(128) void gemm_bt_tdm_kernel(
    const unsigned short* __restrict__ A, const unsigned short* __restrict__ B,
    float* __restrict__ C, const float* __restrict__ bias,
    int M, int N, int K, int lda, int ldb, int ldc, float alpha)
{
    __shared__ __align__(16) unsigned short shA[2][64 * 32];
    __shared__ __align__(16) unsigned short shB[2][64 * 32];

    const int lane = threadIdx.x & 31;
    const int wave = threadIdx.x >> 5;           // 0..3
    const int row0 = blockIdx.y * 64;
    const int col0 = blockIdx.x * 64;
    const int wr = (wave >> 1) * 32;             // wave sub-tile in block tile
    const int wc = (wave & 1) * 32;
    (void)M; (void)N;

    f32x8 acc00 = {}; f32x8 acc01 = {}; f32x8 acc10 = {}; f32x8 acc11 = {};

    // prologue: DMA tile 0 into buffer 0 (wave 0 issues; TDM ignores EXEC)
    if (threadIdx.x < 32) {
        tdm_load_tile(A + (size_t)row0 * lda, ldsOff(&shA[0][0]), 64, 32, lda);
        tdm_load_tile(B + (size_t)col0 * ldb, ldsOff(&shB[0][0]), 64, 32, ldb);
        __builtin_amdgcn_s_wait_tensorcnt(0);
    }
    __syncthreads();

    int buf = 0;
    for (int k0 = 0; k0 < K; k0 += 32) {
        // kick off next tile DMA into the other buffer
        if (threadIdx.x < 32 && k0 + 32 < K) {
            tdm_load_tile(A + (size_t)row0 * lda + (k0 + 32), ldsOff(&shA[buf ^ 1][0]), 64, 32, lda);
            tdm_load_tile(B + (size_t)col0 * ldb + (k0 + 32), ldsOff(&shB[buf ^ 1][0]), 64, 32, ldb);
        }

        const unsigned short* sa = &shA[buf][0];
        const unsigned short* sb = &shB[buf][0];
        bf16x16 a0 = load_frag(sa + (size_t)(wr     ) * 32, 32, lane);
        bf16x16 a1 = load_frag(sa + (size_t)(wr + 16) * 32, 32, lane);
        bf16x16 b0 = load_frag(sb + (size_t)(wc     ) * 32, 32, lane);
        bf16x16 b1 = load_frag(sb + (size_t)(wc + 16) * 32, 32, lane);
        acc00 = __builtin_amdgcn_wmma_f32_16x16x32_bf16(false, a0, false, b0, (short)0, acc00, false, false);
        acc01 = __builtin_amdgcn_wmma_f32_16x16x32_bf16(false, a0, false, b1, (short)0, acc01, false, false);
        acc10 = __builtin_amdgcn_wmma_f32_16x16x32_bf16(false, a1, false, b0, (short)0, acc10, false, false);
        acc11 = __builtin_amdgcn_wmma_f32_16x16x32_bf16(false, a1, false, b1, (short)0, acc11, false, false);

        // wait for the in-flight next tile (overlapped with the WMMAs above), then publish
        if (threadIdx.x < 32) __builtin_amdgcn_s_wait_tensorcnt(0);
        __syncthreads();
        buf ^= 1;
    }

    const int crow = (lane >> 4) * 8;
    const int ccol = lane & 15;
    const float bc0 = bias ? bias[col0 + wc + ccol]      : 0.f;
    const float bc1 = bias ? bias[col0 + wc + 16 + ccol] : 0.f;
    #pragma unroll
    for (int j = 0; j < 8; ++j) {
        int r0 = row0 + wr + crow + j;
        int r1 = r0 + 16;
        C[(size_t)r0 * ldc + col0 + wc + ccol]      = alpha * acc00[j] + bc0;
        C[(size_t)r0 * ldc + col0 + wc + 16 + ccol] = alpha * acc01[j] + bc1;
        C[(size_t)r1 * ldc + col0 + wc + ccol]      = alpha * acc10[j] + bc0;
        C[(size_t)r1 * ldc + col0 + wc + 16 + ccol] = alpha * acc11[j] + bc1;
    }
}

// ---------------- LayerNorm -> bf16(y), bf16(y*y) ----------------
__global__ __launch_bounds__(256) void ln_pack_kernel(
    const float* __restrict__ X, const float* __restrict__ g, const float* __restrict__ be,
    unsigned short* __restrict__ ob, unsigned short* __restrict__ osq)
{
    const int r = blockIdx.x, t = threadIdx.x;
    const float* x = X + (size_t)r * DD_;
    float x0 = x[t], x1 = x[t + 256];
    __shared__ float red[256];
    red[t] = x0 + x1; __syncthreads();
    for (int o = 128; o > 0; o >>= 1) { if (t < o) red[t] += red[t + o]; __syncthreads(); }
    float mu = red[0] * (1.f / DD_); __syncthreads();
    float d0 = x0 - mu, d1 = x1 - mu;
    red[t] = d0 * d0 + d1 * d1; __syncthreads();
    for (int o = 128; o > 0; o >>= 1) { if (t < o) red[t] += red[t + o]; __syncthreads(); }
    float rstd = rsqrtf(red[0] * (1.f / DD_) + 1e-5f);
    float y0 = d0 * rstd * g[t] + be[t];
    float y1 = d1 * rstd * g[t + 256] + be[t + 256];
    size_t base = (size_t)r * DD_;
    ob[base + t] = f2bf(y0);           ob[base + t + 256] = f2bf(y1);
    osq[base + t] = f2bf(y0 * y0);     osq[base + t + 256] = f2bf(y1 * y1);
}

// ---------------- dist2[b,i,j] = max(|xi|^2+|xj|^2-2 xi.xj, 0) ----------------
__global__ __launch_bounds__(256) void dist2_kernel(const float* __restrict__ xr, float* __restrict__ out)
{
    const int bi = blockIdx.x;           // b*1024 + i
    __shared__ float xi[DRAW_];
    if (threadIdx.x < DRAW_) xi[threadIdx.x] = xr[(size_t)bi * DRAW_ + threadIdx.x];
    __syncthreads();
    const int b = bi >> 10;
    for (int j = threadIdx.x; j < NN_; j += 256) {
        const float* xj = xr + ((size_t)(b * NN_ + j)) * DRAW_;
        float dot = 0.f, si = 0.f, sj = 0.f;
        #pragma unroll
        for (int k = 0; k < DRAW_; ++k) { float a = xi[k], c = xj[k]; dot += a * c; si += a * a; sj += c * c; }
        out[(size_t)bi * NN_ + j] = fmaxf(si + sj - 2.f * dot, 0.f);
    }
}

// ---------------- softmax over (scores + RBF) -> bf16 attn, bf16 attn^2 ----------------
__global__ __launch_bounds__(256) void softmax_rbf_kernel(
    const float* __restrict__ S, const float* __restrict__ dist2,
    const float* __restrict__ lsf, const float* __restrict__ llen,
    unsigned short* __restrict__ attn, unsigned short* __restrict__ attnsq,
    int b, int h)
{
    const int i = blockIdx.x, t = threadIdx.x;
    const float sf2    = expf(2.f * lsf[h]);
    const float inv2l2 = 0.5f * expf(-2.f * llen[h]);
    const float* srow = S + (size_t)i * NN_;
    const float* drow = dist2 + ((size_t)(b * NN_ + i)) * NN_;
    float v[4]; float m = -3.4e38f;
    #pragma unroll
    for (int c = 0; c < 4; ++c) {
        int j = t + 256 * c;
        float x = srow[j] + sf2 * expf(-drow[j] * inv2l2);
        v[c] = x; m = fmaxf(m, x);
    }
    __shared__ float red[256];
    red[t] = m; __syncthreads();
    for (int o = 128; o > 0; o >>= 1) { if (t < o) red[t] = fmaxf(red[t], red[t + o]); __syncthreads(); }
    m = red[0]; __syncthreads();
    float sum = 0.f;
    #pragma unroll
    for (int c = 0; c < 4; ++c) { v[c] = expf(v[c] - m); sum += v[c]; }
    red[t] = sum; __syncthreads();
    for (int o = 128; o > 0; o >>= 1) { if (t < o) red[t] += red[t + o]; __syncthreads(); }
    float inv = 1.f / red[0];
    #pragma unroll
    for (int c = 0; c < 4; ++c) {
        int j = t + 256 * c;
        float p = v[c] * inv;
        attn[(size_t)i * NN_ + j] = f2bf(p);
        attnsq[(size_t)i * NN_ + j] = f2bf(p * p);
    }
}

// ---------------- GELU: am, am^2, av = gelu'(h)^2 * hv (all bf16) ----------------
__global__ void gelu_pack_kernel(const float* __restrict__ hm, const float* __restrict__ hv,
                                 unsigned short* __restrict__ amb, unsigned short* __restrict__ amsq,
                                 unsigned short* __restrict__ avb, int n)
{
    int i = blockIdx.x * blockDim.x + threadIdx.x;
    if (i >= n) return;
    float x = hm[i];
    float phi = 0.5f * (1.f + erff(x * 0.70710678f));
    float am = x * phi;
    float pdf = expf(-0.5f * x * x) * 0.39894228f;
    float gd = phi + x * pdf;
    float av = gd * gd * hv[i];
    amb[i] = f2bf(am); amsq[i] = f2bf(am * am); avb[i] = f2bf(av);
}

// ---------------- small elementwise / pack kernels ----------------
__global__ void pack_bf16_kernel(const float* x, unsigned short* o, int n) {
    int i = blockIdx.x * blockDim.x + threadIdx.x;
    if (i < n) o[i] = f2bf(x[i]);
}
__global__ void pack_pair_kernel(const float* x, unsigned short* o, unsigned short* osq, int n) {
    int i = blockIdx.x * blockDim.x + threadIdx.x;
    if (i < n) { float v = x[i]; o[i] = f2bf(v); osq[i] = f2bf(v * v); }
}
__global__ void pack_sig2_bf16_kernel(const float* rho, unsigned short* o, int n) {
    int i = blockIdx.x * blockDim.x + threadIdx.x;
    if (i < n) { float s = softplusf(rho[i]); o[i] = f2bf(s * s); }
}
__global__ void pack_comb_kernel(const float* mu, const float* rho, unsigned short* o, int n) {
    int i = blockIdx.x * blockDim.x + threadIdx.x;
    if (i < n) { float m = mu[i], s = softplusf(rho[i]); o[i] = f2bf(m * m + s * s); }
}
__global__ void sig2_f32_kernel(const float* rho, float* o, int n) {
    int i = blockIdx.x * blockDim.x + threadIdx.x;
    if (i < n) { float s = softplusf(rho[i]); o[i] = s * s; }
}
__global__ void addf_kernel(const float* a, const float* b, float* o, int n) {
    int i = blockIdx.x * blockDim.x + threadIdx.x;
    if (i < n) o[i] = a[i] + b[i];
}
__global__ void copyf_kernel(const float* a, float* o, int n) {
    int i = blockIdx.x * blockDim.x + threadIdx.x;
    if (i < n) o[i] = a[i];
}
// [B*N, D] f32  ->  bf16 [B,H,dk,N] (per-head transpose so GEMM B-operand is K-contiguous)
__global__ void transpose_pack_kernel(const float* __restrict__ x, unsigned short* __restrict__ o, int n) {
    int idx = blockIdx.x * blockDim.x + threadIdx.x;
    if (idx >= n) return;
    int row = idx >> 9;            // /512
    int col = idx & 511;
    int b = row >> 10, nn = row & 1023;
    int h = col >> 6, dd = col & 63;
    o[(((size_t)(b * NH_ + h) * DK_ + dd) << 10) + nn] = f2bf(x[idx]);
}

// ---------------- cascade combine + early-exit machinery ----------------
__global__ void combine_kernel(const float* __restrict__ Hc, const float* __restrict__ Hvc,
                               const float* __restrict__ Hb, const float* __restrict__ fm,
                               const float* __restrict__ ov, const float* __restrict__ vta,
                               const float* __restrict__ ovd, const float* __restrict__ vc,
                               const float* __restrict__ lg, const float* __restrict__ flag,
                               float* __restrict__ Ho, float* __restrict__ Hvo, int n)
{
    int i = blockIdx.x * blockDim.x + threadIdx.x;
    if (i >= n) return;
    float g1 = softplusf(lg[0]), g2 = softplusf(lg[1]);
    float g3 = softplusf(lg[2]), g4 = softplusf(lg[3]);
    float alpha = g1 / (g1 + g2), beta = g3 / (g3 + g4);
    float Va = g1 * g2 / ((g1 + g2) * (g1 + g2) * (g1 + g2 + 1.f));
    float Vb = g3 * g4 / ((g3 + g4) * (g3 + g4) * (g3 + g4 + 1.f));
    float H = Hc[i], Hv = Hvc[i];
    float F = Hb[i] + fm[i];
    float Fv = ov[i] + vta[i] + ovd[i] + vc[i];
    float Hn = alpha * H + beta * F;
    float Hvn = alpha * alpha * Hv + beta * beta * Fv + Va * H * H + Vb * F * F + Va * Hv + Vb * Fv;
    bool done = flag[0] > 0.5f;
    Ho[i]  = done ? H  : Hn;
    Hvo[i] = done ? Hv : Hvn;
}
__global__ void init_flags_kernel(float* accum, float* flag, int clearFlag) {
    accum[0] = 0.f;
    if (clearFlag) flag[0] = 0.f;
}
__global__ __launch_bounds__(256) void reduce_sum_kernel(const float* __restrict__ x, float* accum, int n) {
    __shared__ float s[256];
    float v = 0.f;
    for (int i = blockIdx.x * blockDim.x + threadIdx.x; i < n; i += gridDim.x * blockDim.x) v += x[i];
    s[threadIdx.x] = v; __syncthreads();
    for (int o = 128; o > 0; o >>= 1) { if (threadIdx.x < o) s[threadIdx.x] += s[threadIdx.x + o]; __syncthreads(); }
    if (threadIdx.x == 0) atomicAdd(accum, s[0]);
}
__global__ void flag_update_kernel(const float* accum, const float* tau, float* flag, float invn) {
    if (flag[0] > 0.5f) return;
    if (accum[0] * invn <= tau[0]) flag[0] = 1.f;
}

// ---------------- workspace layout ----------------
constexpr size_t A256(size_t x) { return (x + 255) & ~(size_t)255; }
constexpr size_t E_BND  = (size_t)BNROWS * DD_;     // 2,097,152
constexpr size_t E_BNF  = (size_t)BNROWS * DFF_;    // 8,388,608
constexpr size_t E_NN   = (size_t)NN_ * NN_;        // 1,048,576
constexpr size_t E_DDm  = (size_t)DD_ * DD_;
constexpr size_t E_FD   = (size_t)DFF_ * DD_;

constexpr size_t o_n1b   = 0;
constexpr size_t o_n1sq  = o_n1b   + A256(E_BND * 2);
constexpr size_t o_Qb    = o_n1sq  + A256(E_BND * 2);
constexpr size_t o_Kb    = o_Qb    + A256(E_BND * 2);
constexpr size_t o_Vt    = o_Kb    + A256(E_BND * 2);
constexpr size_t o_Vvt   = o_Vt    + A256(E_BND * 2);
constexpr size_t o_ctxb  = o_Vvt   + A256(E_BND * 2);
constexpr size_t o_ctxsq = o_ctxb  + A256(E_BND * 2);
constexpr size_t o_n2b   = o_ctxsq + A256(E_BND * 2);
constexpr size_t o_n2sq  = o_n2b   + A256(E_BND * 2);
constexpr size_t o_amb   = o_n2sq  + A256(E_BND * 2);
constexpr size_t o_amsq  = o_amb   + A256(E_BNF * 2);
constexpr size_t o_avb   = o_amsq  + A256(E_BNF * 2);
constexpr size_t o_Ab    = o_avb   + A256(E_BNF * 2);
constexpr size_t o_Asqb  = o_Ab    + A256(E_NN * 2);
constexpr size_t o_Wq    = o_Asqb  + A256(E_NN * 2);
constexpr size_t o_Wk    = o_Wq    + A256(E_DDm * 2);
constexpr size_t o_Wv    = o_Wk    + A256(E_DDm * 2);
constexpr size_t o_Wo    = o_Wv    + A256(E_DDm * 2);
constexpr size_t o_Wvs2  = o_Wo    + A256(E_DDm * 2);
constexpr size_t o_Wos2  = o_Wvs2  + A256(E_DDm * 2);
constexpr size_t o_W1b   = o_Wos2  + A256(E_DDm * 2);
constexpr size_t o_W1s2  = o_W1b   + A256(E_FD * 2);
constexpr size_t o_W2b   = o_W1s2  + A256(E_FD * 2);
constexpr size_t o_W2s2  = o_W2b   + A256(E_FD * 2);
constexpr size_t o_W2c   = o_W2s2  + A256(E_FD * 2);
constexpr size_t o_Cf    = o_W2c   + A256(E_FD * 2);     // generic f32 GEMM out (also hm)
constexpr size_t o_hv    = o_Cf    + A256(E_BNF * 4);
constexpr size_t o_dist2 = o_hv    + A256(E_BNF * 4);
constexpr size_t o_Sh    = o_dist2 + A256((size_t)BB * E_NN * 4);
constexpr size_t o_ctx   = o_Sh    + A256(E_NN * 4);
constexpr size_t o_vta   = o_ctx   + A256(E_BND * 4);
constexpr size_t o_ov    = o_vta   + A256(E_BND * 4);
constexpr size_t o_om    = o_ov    + A256(E_BND * 4);
constexpr size_t o_Hbres = o_om    + A256(E_BND * 4);
constexpr size_t o_fm    = o_Hbres + A256(E_BND * 4);
constexpr size_t o_ovd   = o_fm    + A256(E_BND * 4);
constexpr size_t o_vc    = o_ovd   + A256(E_BND * 4);
constexpr size_t o_Hst0  = o_vc    + A256(E_BND * 4);
constexpr size_t o_Hvst0 = o_Hst0  + A256(E_BND * 4);
constexpr size_t o_Hst1  = o_Hvst0 + A256(E_BND * 4);
constexpr size_t o_Hvst1 = o_Hst1  + A256(E_BND * 4);
constexpr size_t o_bvs2  = o_Hvst1 + A256(E_BND * 4);
constexpr size_t o_bos2  = o_bvs2  + A256(DD_ * 4);
constexpr size_t o_b1s2  = o_bos2  + A256(DD_ * 4);
constexpr size_t o_b2s2  = o_b1s2  + A256(DFF_ * 4);
constexpr size_t o_accum = o_b2s2  + A256(DD_ * 4);
constexpr size_t o_flag  = o_accum + 256;

// ---------------- launch ----------------
extern "C" void kernel_launch(void* const* d_in, const int* in_sizes, int n_in,
                              void* d_out, int out_size, void* d_ws, size_t ws_size,
                              hipStream_t stream)
{
    (void)in_sizes; (void)n_in; (void)out_size; (void)ws_size;
    char* w = (char*)d_ws;
    auto US  = [&](size_t o) { return (unsigned short*)(w + o); };
    auto F32 = [&](size_t o) { return (float*)(w + o); };

    const float* H_in   = (const float*)d_in[0];
    const float* Hv_in  = (const float*)d_in[1];
    const float* x_raw  = (const float*)d_in[2];
    const float* tau    = (const float*)d_in[3];
    const float* aw_mu  = (const float*)d_in[4];
    const float* aw_rho = (const float*)d_in[5];
    const float* ab_mu  = (const float*)d_in[6];
    const float* ab_rho = (const float*)d_in[7];
    const float* w1_mu  = (const float*)d_in[8];
    const float* w1_rho = (const float*)d_in[9];
    const float* b1_mu  = (const float*)d_in[10];
    const float* b1_rho = (const float*)d_in[11];
    const float* w2_mu  = (const float*)d_in[12];
    const float* w2_rho = (const float*)d_in[13];
    const float* b2_mu  = (const float*)d_in[14];
    const float* b2_rho = (const float*)d_in[15];
    const float* ln1_g  = (const float*)d_in[16];
    const float* ln1_b  = (const float*)d_in[17];
    const float* ln2_g  = (const float*)d_in[18];
    const float* ln2_b  = (const float*)d_in[19];
    const float* lsf    = (const float*)d_in[20];
    const float* llen   = (const float*)d_in[21];
    const float* lgam   = (const float*)d_in[22];

    // direct-load WMMA GEMM (small K)
    auto gemm = [&](const unsigned short* A, const unsigned short* B, float* C, const float* bias,
                    int M, int N, int K, int lda, int ldb, int ldc, float alpha) {
        dim3 g(N / 64, M / 64);
        gemm_bt_kernel<<<g, dim3(128), 0, stream>>>(A, B, C, bias, M, N, K, lda, ldb, ldc, alpha);
    };
    // TDM + LDS double-buffered WMMA GEMM (large K)
    auto gemmT = [&](const unsigned short* A, const unsigned short* B, float* C, const float* bias,
                     int M, int N, int K, int lda, int ldb, int ldc, float alpha) {
        dim3 g(N / 64, M / 64);
        gemm_bt_tdm_kernel<<<g, dim3(128), 0, stream>>>(A, B, C, bias, M, N, K, lda, ldb, ldc, alpha);
    };
    auto eltGrid = [](int n) { return dim3((n + 255) / 256); };

    // ---- one-time setup (recomputed every launch; deterministic) ----
    init_flags_kernel<<<1, 1, 0, stream>>>(F32(o_accum), F32(o_flag), 1);
    copyf_kernel<<<eltGrid(E_BND), 256, 0, stream>>>(H_in,  F32(o_Hst0),  (int)E_BND);
    copyf_kernel<<<eltGrid(E_BND), 256, 0, stream>>>(Hv_in, F32(o_Hvst0), (int)E_BND);

    // weight packs
    pack_bf16_kernel<<<eltGrid(E_DDm), 256, 0, stream>>>(aw_mu + 0 * E_DDm, US(o_Wq), (int)E_DDm);
    pack_bf16_kernel<<<eltGrid(E_DDm), 256, 0, stream>>>(aw_mu + 1 * E_DDm, US(o_Wk), (int)E_DDm);
    pack_bf16_kernel<<<eltGrid(E_DDm), 256, 0, stream>>>(aw_mu + 2 * E_DDm, US(o_Wv), (int)E_DDm);
    pack_bf16_kernel<<<eltGrid(E_DDm), 256, 0, stream>>>(aw_mu + 3 * E_DDm, US(o_Wo), (int)E_DDm);
    pack_sig2_bf16_kernel<<<eltGrid(E_DDm), 256, 0, stream>>>(aw_rho + 2 * E_DDm, US(o_Wvs2), (int)E_DDm);
    pack_sig2_bf16_kernel<<<eltGrid(E_DDm), 256, 0, stream>>>(aw_rho + 3 * E_DDm, US(o_Wos2), (int)E_DDm);
    pack_bf16_kernel<<<eltGrid(E_FD), 256, 0, stream>>>(w1_mu,  US(o_W1b),  (int)E_FD);
    pack_sig2_bf16_kernel<<<eltGrid(E_FD), 256, 0, stream>>>(w1_rho, US(o_W1s2), (int)E_FD);
    pack_bf16_kernel<<<eltGrid(E_FD), 256, 0, stream>>>(w2_mu,  US(o_W2b),  (int)E_FD);
    pack_sig2_bf16_kernel<<<eltGrid(E_FD), 256, 0, stream>>>(w2_rho, US(o_W2s2), (int)E_FD);
    pack_comb_kernel<<<eltGrid(E_FD), 256, 0, stream>>>(w2_mu, w2_rho, US(o_W2c), (int)E_FD);
    sig2_f32_kernel<<<eltGrid(DD_), 256, 0, stream>>>(ab_rho + 2 * DD_, F32(o_bvs2), DD_);
    sig2_f32_kernel<<<eltGrid(DD_), 256, 0, stream>>>(ab_rho + 3 * DD_, F32(o_bos2), DD_);
    sig2_f32_kernel<<<eltGrid(DFF_), 256, 0, stream>>>(b1_rho, F32(o_b1s2), DFF_);
    sig2_f32_kernel<<<eltGrid(DD_), 256, 0, stream>>>(b2_rho, F32(o_b2s2), DD_);

    // RBF distance matrix (depends only on x_raw)
    dist2_kernel<<<BB * NN_, 256, 0, stream>>>(x_raw, F32(o_dist2));

    float* Hc  = F32(o_Hst0);  float* Hvc  = F32(o_Hvst0);
    float* Hnx = F32(o_Hst1);  float* Hvnx = F32(o_Hvst1);

    for (int it = 0; it < TMAX_; ++it) {
        // ---- attention block ----
        ln_pack_kernel<<<BNROWS, 256, 0, stream>>>(Hc, ln1_g, ln1_b, US(o_n1b), US(o_n1sq));

        // qm / km / vm / vv  (vv uses x^2 vs sig^2)
        gemmT(US(o_n1b), US(o_Wq), F32(o_Cf), ab_mu + 0 * DD_, BNROWS, DD_, DD_, DD_, DD_, DD_, 1.f);
        pack_bf16_kernel<<<eltGrid(E_BND), 256, 0, stream>>>(F32(o_Cf), US(o_Qb), (int)E_BND);
        gemmT(US(o_n1b), US(o_Wk), F32(o_Cf), ab_mu + 1 * DD_, BNROWS, DD_, DD_, DD_, DD_, DD_, 1.f);
        pack_bf16_kernel<<<eltGrid(E_BND), 256, 0, stream>>>(F32(o_Cf), US(o_Kb), (int)E_BND);
        gemmT(US(o_n1b), US(o_Wv), F32(o_Cf), ab_mu + 2 * DD_, BNROWS, DD_, DD_, DD_, DD_, DD_, 1.f);
        transpose_pack_kernel<<<eltGrid(E_BND), 256, 0, stream>>>(F32(o_Cf), US(o_Vt), (int)E_BND);
        gemmT(US(o_n1sq), US(o_Wvs2), F32(o_Cf), F32(o_bvs2), BNROWS, DD_, DD_, DD_, DD_, DD_, 1.f);
        transpose_pack_kernel<<<eltGrid(E_BND), 256, 0, stream>>>(F32(o_Cf), US(o_Vvt), (int)E_BND);

        // per-(batch, head): scores -> softmax(+RBF) -> attn@V, attn^2@Vv
        for (int bh = 0; bh < BB * NH_; ++bh) {
            int b = bh >> 3, h = bh & 7;
            const unsigned short* Qh = US(o_Qb) + (size_t)(b * NN_) * DD_ + h * DK_;
            const unsigned short* Kh = US(o_Kb) + (size_t)(b * NN_) * DD_ + h * DK_;
            gemm(Qh, Kh, F32(o_Sh), nullptr, NN_, NN_, DK_, DD_, DD_, NN_, 0.125f);  // 1/sqrt(64)
            softmax_rbf_kernel<<<NN_, 256, 0, stream>>>(F32(o_Sh), F32(o_dist2), lsf, llen,
                                                        US(o_Ab), US(o_Asqb), b, h);
            const unsigned short* Vth  = US(o_Vt)  + (size_t)bh * DK_ * NN_;
            const unsigned short* Vvth = US(o_Vvt) + (size_t)bh * DK_ * NN_;
            float* ctxh = F32(o_ctx) + (size_t)(b * NN_) * DD_ + h * DK_;
            float* vtah = F32(o_vta) + (size_t)(b * NN_) * DD_ + h * DK_;
            gemmT(US(o_Ab),   Vth,  ctxh, nullptr, NN_, DK_, NN_, NN_, NN_, DD_, 1.f);
            gemmT(US(o_Asqb), Vvth, vtah, nullptr, NN_, DK_, NN_, NN_, NN_, DD_, 1.f);
        }

        // output projection: mean + variance path
        pack_pair_kernel<<<eltGrid(E_BND), 256, 0, stream>>>(F32(o_ctx), US(o_ctxb), US(o_ctxsq), (int)E_BND);
        gemmT(US(o_ctxb),  US(o_Wo),   F32(o_om), ab_mu + 3 * DD_, BNROWS, DD_, DD_, DD_, DD_, DD_, 1.f);
        gemmT(US(o_ctxsq), US(o_Wos2), F32(o_ov), F32(o_bos2),     BNROWS, DD_, DD_, DD_, DD_, DD_, 1.f);

        // residual + second LN
        addf_kernel<<<eltGrid(E_BND), 256, 0, stream>>>(Hc, F32(o_om), F32(o_Hbres), (int)E_BND);
        ln_pack_kernel<<<BNROWS, 256, 0, stream>>>(F32(o_Hbres), ln2_g, ln2_b, US(o_n2b), US(o_n2sq));

        // ---- FFN ----
        gemmT(US(o_n2b),  US(o_W1b),  F32(o_Cf), b1_mu,       BNROWS, DFF_, DD_, DD_, DD_, DFF_, 1.f);  // hm
        gemmT(US(o_n2sq), US(o_W1s2), F32(o_hv), F32(o_b1s2), BNROWS, DFF_, DD_, DD_, DD_, DFF_, 1.f);  // hv
        gelu_pack_kernel<<<eltGrid(E_BNF), 256, 0, stream>>>(F32(o_Cf), F32(o_hv),
                                                             US(o_amb), US(o_amsq), US(o_avb), (int)E_BNF);
        gemmT(US(o_amb),  US(o_W2b),  F32(o_fm),  b2_mu,       BNROWS, DD_, DFF_, DFF_, DFF_, DD_, 1.f);
        gemmT(US(o_amsq), US(o_W2s2), F32(o_ovd), F32(o_b2s2), BNROWS, DD_, DFF_, DFF_, DFF_, DD_, 1.f);
        gemmT(US(o_avb),  US(o_W2c),  F32(o_vc),  nullptr,     BNROWS, DD_, DFF_, DFF_, DFF_, DD_, 1.f);

        // ---- cascade combine + early-exit flag ----
        combine_kernel<<<eltGrid(E_BND), 256, 0, stream>>>(
            Hc, Hvc, F32(o_Hbres), F32(o_fm), F32(o_ov), F32(o_vta), F32(o_ovd), F32(o_vc),
            lgam, F32(o_flag), Hnx, Hvnx, (int)E_BND);
        init_flags_kernel<<<1, 1, 0, stream>>>(F32(o_accum), F32(o_flag), 0);
        reduce_sum_kernel<<<1024, 256, 0, stream>>>(Hvnx, F32(o_accum), (int)E_BND);
        flag_update_kernel<<<1, 1, 0, stream>>>(F32(o_accum), tau, F32(o_flag), 1.f / (float)E_BND);

        // swap state buffers
        float* t0 = Hc;  Hc = Hnx;   Hnx = t0;
        float* t1 = Hvc; Hvc = Hvnx; Hvnx = t1;
    }

    // outputs: (H, H_var) concatenated
    float* out = (float*)d_out;
    copyf_kernel<<<eltGrid(E_BND), 256, 0, stream>>>(Hc,  out,                 (int)E_BND);
    copyf_kernel<<<eltGrid(E_BND), 256, 0, stream>>>(Hvc, out + E_BND,         (int)E_BND);
}
